// Mamba_79757542686750
// MI455X (gfx1250) — matrix operand order repaired and checked by reading, not compiled, not run on hardware
//
#include <hip/hip_runtime.h>
#include <hip/hip_bf16.h>

// ---------------------------------------------------------------------------
// Mamba forward (B=1, L=2048, D_MODEL=768, N_LAYER=4, D_INNER=1536,
// D_STATE=16, DT_RANK=48, VOCAB=50257) for MI455X (gfx1250, wave32).
//
// GEMMs: LDS-tiled bf16 WMMA (V_WMMA_F32_16X16X32_BF16):
//   - 64(M) x 128(N) block tile, K staged 32 at a time, fp32 in LDS
//   - FULL (interior) blocks stage via GLOBAL_LOAD_ASYNC_TO_LDS_B128
//     (ASYNCcnt-tracked, s_wait_asynccnt) -- no VGPR round trip
//   - double-buffered LDS ping-pong, one barrier per K-step
//   - padded LDS rows (32 -> 40 fp32) for aligned v8f fragment reads
//   - bf16 conversion at fragment build (v_cvt_pk_bf16_f32, overlaps WMMA)
//   - guarded instantiation for M/N/K tail blocks (plain loads/stores)
// ---------------------------------------------------------------------------

#define MAMBA_L      2048
#define MAMBA_DM     768
#define MAMBA_DI     1536
#define MAMBA_DX     3072   // 2*D_INNER
#define MAMBA_DST    16
#define MAMBA_DTR    48
#define MAMBA_NX     80     // DT_RANK + 2*D_STATE
#define MAMBA_NL     4
#define MAMBA_VOC    50257

typedef __bf16 bf16_t;
typedef bf16_t v16bf __attribute__((ext_vector_type(16)));
typedef float  v8f   __attribute__((ext_vector_type(8)));
typedef float  v4f   __attribute__((ext_vector_type(4)));

__device__ __forceinline__ float silu_f(float x) {
    return x / (1.0f + __expf(-x));
}
__device__ __forceinline__ float softplus_f(float x) {
    return (x > 20.0f) ? x : log1pf(__expf(x));
}

// --- CDNA5 async global->LDS copy (16 bytes / lane), ASYNCcnt-tracked ------
__device__ __forceinline__ void async_g2l_b128(unsigned lds_byte_off,
                                               const float* gptr)
{
    asm volatile("global_load_async_to_lds_b128 %0, %1, off"
                 :: "v"(lds_byte_off), "v"(gptr)
                 : "memory");
}
__device__ __forceinline__ void wait_async0()
{
    asm volatile("s_wait_asynccnt 0x0" ::: "memory");
}
__device__ __forceinline__ unsigned lds_off_u32(const void* p)
{
    // generic LDS pointers carry the LDS byte offset in the low 32 bits
    return (unsigned)(unsigned long long)p;
}

// ---------------------------------------------------------------------------
// GEMM: C[M,N] = act( A[M,K](lda) @ W[N,K]^T + bias[n] ) + residual
// ---------------------------------------------------------------------------
#define BLK_M 64
#define BLK_N 128
#define BLK_K 32
#define LDS_SF 40                      // padded LDS row stride (fp32 elems)
#define ASZF  (BLK_M * LDS_SF)         // 2560 floats / A buffer (10 KB)
#define BSZF  (BLK_N * LDS_SF)         // 5120 floats / B buffer (20 KB)

template <bool FULL>
__device__ __forceinline__ void gemm_core(
    const float* __restrict__ A, int lda,
    const float* __restrict__ W,
    const float* __restrict__ bias,
    const float* __restrict__ residual,
    float* __restrict__ C,
    int M, int N, int K, int act,
    int m0, int n0, float* ldsA, float* ldsB)
{
    const int tid  = threadIdx.x;
    const int lane = tid & 31;
    const int wave = tid >> 5;
    const int wm   = wave & 3;    // 16-row M slice
    const int wn   = wave >> 2;   // 64-col N slice

    const int r    = lane & 15;
    const int half = lane >> 4;
    const int kbA  = half << 4;   // 0 or 16
    const int kbB  = half << 4;   // 0 or 16

    // stage one K tile (A: 2 rounds of b128, B: 4 rounds) into buffer `buf`
    auto stage = [&](int k0, int buf) {
        float* lA = ldsA + buf * ASZF;
        float* lB = ldsB + buf * BSZF;
        // A tile: 64 rows x 32 K, unit = 4 floats
        #pragma unroll
        for (int it = 0; it < 2; ++it) {
            const int u   = tid + (it << 8);   // 0..511
            const int row = u >> 3;            // 0..63
            const int seg = (u & 7) << 2;      // 0,4,...,28
            const float* p = A + (size_t)(m0 + row) * lda + (k0 + seg);
            float* q = &lA[row * LDS_SF + seg];
            if (FULL) {
                async_g2l_b128(lds_off_u32(q), p);
            } else {
                v4f f = {};
                if ((m0 + row) < M && (k0 + seg + 4) <= K)
                    f = *(const v4f*)(p);
                *(v4f*)q = f;
            }
        }
        // B tile: 128 rows x 32 K
        #pragma unroll
        for (int it = 0; it < 4; ++it) {
            const int u   = tid + (it << 8);   // 0..1023
            const int row = u >> 3;            // 0..127
            const int seg = (u & 7) << 2;
            const float* p = W + (size_t)(n0 + row) * K + (k0 + seg);
            float* q = &lB[row * LDS_SF + seg];
            if (FULL) {
                async_g2l_b128(lds_off_u32(q), p);
            } else {
                v4f f = {};
                if ((n0 + row) < N && (k0 + seg + 4) <= K)
                    f = *(const v4f*)(p);
                *(v4f*)q = f;
            }
        }
    };

    v8f acc[4] = {};
    const int nk = (K + BLK_K - 1) / BLK_K;

    stage(0, 0);   // prologue: tile 0 -> buffer 0

    for (int kt = 0; kt < nk; ++kt) {
        if (FULL) wait_async0();   // my async writes to LDS are complete
        __syncthreads();           // everyone's writes visible

        const int  cur     = kt & 1;
        const bool hasNext = (kt + 1) < nk;
        if (hasNext) stage((kt + 1) * BLK_K, cur ^ 1);  // overlap w/ compute

        // ---- compute from LDS buffer `cur` ----
        const float* lA = ldsA + cur * ASZF;
        const float* lB = ldsB + cur * BSZF;

        // A fragment (16x32): e0..15 -> K=kbA+e, contiguous
        const int abase = (wm * 16 + r) * LDS_SF;
        const v8f a0 = *(const v8f*)&lA[abase + kbA];
        const v8f a1 = *(const v8f*)&lA[abase + kbA + 8];
        v16bf afrag;
        #pragma unroll
        for (int j = 0; j < 8; ++j) {
            afrag[j]     = (bf16_t)a0[j];
            afrag[8 + j] = (bf16_t)a1[j];
        }

        // B fragments (32x16 each): e0..15 -> K=kbB+e, contiguous
        v16bf bfrag[4];
        #pragma unroll
        for (int j = 0; j < 4; ++j) {
            const int nb = (wn * 64 + j * 16 + r) * LDS_SF + kbB;
            const v8f b0 = *(const v8f*)&lB[nb];
            const v8f b1 = *(const v8f*)&lB[nb + 8];
            #pragma unroll
            for (int e = 0; e < 8; ++e) {
                bfrag[j][e]     = (bf16_t)b0[e];
                bfrag[j][8 + e] = (bf16_t)b1[e];
            }
        }
        #pragma unroll
        for (int j = 0; j < 4; ++j) {
            acc[j] = __builtin_amdgcn_wmma_f32_16x16x32_bf16(
                         false, afrag, false, bfrag[j], (short)0, acc[j],
                         false, false);
        }
    }

    // ---- epilogue: C/D layout, lane = N col, VGPR v = row v + 8*half ----
    #pragma unroll
    for (int j = 0; j < 4; ++j) {
        const int cn = n0 + wn * 64 + j * 16 + r;
        if (FULL || cn < N) {
            const float bv = bias ? bias[cn] : 0.0f;
            #pragma unroll
            for (int v = 0; v < 8; ++v) {
                const int cm = m0 + wm * 16 + v + (half << 3);
                if (FULL || cm < M) {
                    float val = acc[j][v] + bv;
                    if (act == 1) val = softplus_f(val);
                    if (residual) val += residual[(size_t)cm * N + cn];
                    C[(size_t)cm * N + cn] = val;
                }
            }
        }
    }
}

__global__ __launch_bounds__(256)
void gemm_bf16_wmma(const float* __restrict__ A, int lda,
                    const float* __restrict__ W,
                    const float* __restrict__ bias,
                    const float* __restrict__ residual,
                    float* __restrict__ C,
                    int M, int N, int K, int act)
{
    __shared__ __align__(32) float ldsA[2 * ASZF];   // 20 KB
    __shared__ __align__(32) float ldsB[2 * BSZF];   // 40 KB

    const int tilesN = (N + BLK_N - 1) / BLK_N;
    const int m0 = (blockIdx.x / tilesN) * BLK_M;
    const int n0 = (blockIdx.x % tilesN) * BLK_N;

    const bool full = (m0 + BLK_M <= M) && (n0 + BLK_N <= N) &&
                      (K % BLK_K == 0);
    if (full)
        gemm_core<true >(A, lda, W, bias, residual, C, M, N, K, act,
                         m0, n0, ldsA, ldsB);
    else
        gemm_core<false>(A, lda, W, bias, residual, C, M, N, K, act,
                         m0, n0, ldsA, ldsB);
}

// ---------------------------------------------------------------------------
// Embedding gather: x[l, :] = embedding[idx[l], :]
// ---------------------------------------------------------------------------
__global__ void embed_kernel(const int* __restrict__ idx,
                             const float* __restrict__ emb,
                             float* __restrict__ x, int D)
{
    const int l = blockIdx.x;
    const int t = idx[l];
    const float* src = emb + (size_t)t * D;
    float* dst = x + (size_t)l * D;
    for (int d = threadIdx.x; d < D; d += blockDim.x) dst[d] = src[d];
}

// ---------------------------------------------------------------------------
// Block reductions (wave32 shuffles + LDS)
// ---------------------------------------------------------------------------
__device__ __forceinline__ float block_reduce_sum(float v, float* red)
{
    const int lane = threadIdx.x & 31;
    const int w    = threadIdx.x >> 5;
    const int nw   = blockDim.x >> 5;
    #pragma unroll
    for (int o = 16; o > 0; o >>= 1) v += __shfl_down(v, o);
    if (lane == 0) red[w] = v;
    __syncthreads();
    float r = (threadIdx.x < (unsigned)nw) ? red[threadIdx.x] : 0.0f;
    if (w == 0) {
        #pragma unroll
        for (int o = 16; o > 0; o >>= 1) r += __shfl_down(r, o);
        if (lane == 0) red[0] = r;
    }
    __syncthreads();
    const float out = red[0];
    __syncthreads();
    return out;
}

__device__ __forceinline__ float block_reduce_max(float v, float* red)
{
    const int lane = threadIdx.x & 31;
    const int w    = threadIdx.x >> 5;
    const int nw   = blockDim.x >> 5;
    #pragma unroll
    for (int o = 16; o > 0; o >>= 1) v = fmaxf(v, __shfl_down(v, o));
    if (lane == 0) red[w] = v;
    __syncthreads();
    float r = (threadIdx.x < (unsigned)nw) ? red[threadIdx.x] : -3.4e38f;
    if (w == 0) {
        #pragma unroll
        for (int o = 16; o > 0; o >>= 1) r = fmaxf(r, __shfl_down(r, o));
        if (lane == 0) red[0] = r;
    }
    __syncthreads();
    const float out = red[0];
    __syncthreads();
    return out;
}

// ---------------------------------------------------------------------------
// RMSNorm: y[row,:] = x[row,:] * rsqrt(mean(x^2) + 1e-5) * w
// ---------------------------------------------------------------------------
__global__ void rmsnorm_kernel(const float* __restrict__ x,
                               const float* __restrict__ w,
                               float* __restrict__ y, int D)
{
    __shared__ float red[32];
    const int row = blockIdx.x;
    const float* xr = x + (size_t)row * D;
    float s = 0.0f;
    for (int d = threadIdx.x; d < D; d += blockDim.x) {
        const float v = xr[d];
        s += v * v;
    }
    const float tot = block_reduce_sum(s, red);
    const float inv = rsqrtf(tot / (float)D + 1e-5f);
    float* yr = y + (size_t)row * D;
    for (int d = threadIdx.x; d < D; d += blockDim.x)
        yr[d] = xr[d] * inv * w[d];
}

// ---------------------------------------------------------------------------
// Causal depthwise conv1d (K=4) + SiLU on the xi half of xz (row stride ldxz)
// ---------------------------------------------------------------------------
__global__ void conv_silu_kernel(const float* __restrict__ xz,
                                 const float* __restrict__ w,
                                 const float* __restrict__ b,
                                 float* __restrict__ xc,
                                 int L, int Din, int ldxz)
{
    const int i = blockIdx.x * blockDim.x + threadIdx.x;
    if (i >= L * Din) return;
    const int l = i / Din;
    const int d = i - l * Din;
    float acc = b[d];
    #pragma unroll
    for (int k = 0; k < 4; ++k) {
        const int ls = l - 3 + k;
        if (ls >= 0) acc += w[d * 4 + k] * xz[(size_t)ls * ldxz + d];
    }
    xc[i] = silu_f(acc);
}

// ---------------------------------------------------------------------------
// Selective scan, one thread per channel d, h[16] in registers; z-gate fused
// ---------------------------------------------------------------------------
__global__ void scan_kernel(const float* __restrict__ xc,
                            const float* __restrict__ dt,
                            const float* __restrict__ xdbl,
                            const float* __restrict__ a_log,
                            const float* __restrict__ Dp,
                            const float* __restrict__ xz,
                            float* __restrict__ y,
                            int L, int Din, int ldxdbl, int ldxz)
{
    const int d = blockIdx.x * blockDim.x + threadIdx.x;
    if (d >= Din) return;

    float A[MAMBA_DST], h[MAMBA_DST];
    #pragma unroll
    for (int s = 0; s < MAMBA_DST; ++s) {
        A[s] = -__expf(a_log[(size_t)d * MAMBA_DST + s]);
        h[s] = 0.0f;
    }
    const float Dv = Dp[d];

    for (int t = 0; t < L; ++t) {
        const float dtv = dt[(size_t)t * Din + d];
        const float xv  = xc[(size_t)t * Din + d];
        const float* row = xdbl + (size_t)t * ldxdbl;
        const float dx = dtv * xv;
        float out = 0.0f;
        #pragma unroll
        for (int s = 0; s < MAMBA_DST; ++s) {
            const float Bv = row[MAMBA_DTR + s];
            const float Cv = row[MAMBA_DTR + MAMBA_DST + s];
            h[s] = __expf(dtv * A[s]) * h[s] + dx * Bv;
            out += h[s] * Cv;
        }
        out += xv * Dv;
        const float zv = xz[(size_t)t * ldxz + Din + d];
        y[(size_t)t * Din + d] = out * silu_f(zv);
    }
}

// ---------------------------------------------------------------------------
// Row-wise NLL from logits (log-softmax + gather of target)
// ---------------------------------------------------------------------------
__global__ void nll_kernel(const float* __restrict__ logits,
                           const int* __restrict__ targets,
                           float* __restrict__ nll, int N)
{
    __shared__ float red[32];
    const int row = blockIdx.x;
    const float* lg = logits + (size_t)row * N;

    float mx = -3.4e38f;
    for (int i = threadIdx.x; i < N; i += blockDim.x) mx = fmaxf(mx, lg[i]);
    const float rowmax = block_reduce_max(mx, red);

    float se = 0.0f;
    for (int i = threadIdx.x; i < N; i += blockDim.x)
        se += __expf(lg[i] - rowmax);
    const float rowsum = block_reduce_sum(se, red);

    if (threadIdx.x == 0) {
        const int t  = targets[row];
        int tc = t < 0 ? 0 : t;
        if (tc > N - 1) tc = N - 1;
        const float lp = lg[tc] - rowmax - __logf(rowsum);
        nll[row] = (t >= 0) ? -lp : 0.0f;
    }
}

// ---------------------------------------------------------------------------
// Final masked-mean loss reduction over all rows
// ---------------------------------------------------------------------------
__global__ void loss_kernel(const float* __restrict__ nll,
                            const int* __restrict__ targets,
                            float* __restrict__ out, int R)
{
    __shared__ float red[32];
    float s = 0.0f, mc = 0.0f;
    for (int i = threadIdx.x; i < R; i += blockDim.x) {
        s  += nll[i];
        mc += (targets[i] >= 0) ? 1.0f : 0.0f;
    }
    const float ts = block_reduce_sum(s, red);
    const float tm = block_reduce_sum(mc, red);
    if (threadIdx.x == 0) out[0] = ts / fmaxf(tm, 1.0f);
}

// ---------------------------------------------------------------------------
// Host orchestration
// ---------------------------------------------------------------------------
extern "C" void kernel_launch(void* const* d_in, const int* in_sizes, int n_in,
                              void* d_out, int out_size, void* d_ws, size_t ws_size,
                              hipStream_t stream)
{
    (void)in_sizes; (void)n_in; (void)out_size; (void)ws_size;

    const int*   idx     = (const int*)  d_in[0];
    const int*   tgt     = (const int*)  d_in[1];
    const float* emb     = (const float*)d_in[2];
    const float* norm_w  = (const float*)d_in[3];
    const float* in_w    = (const float*)d_in[4];
    const float* conv_w  = (const float*)d_in[5];
    const float* conv_b  = (const float*)d_in[6];
    const float* xp_w    = (const float*)d_in[7];
    const float* dt_w    = (const float*)d_in[8];
    const float* dt_b    = (const float*)d_in[9];
    const float* a_log   = (const float*)d_in[10];
    const float* d_par   = (const float*)d_in[11];
    const float* out_w   = (const float*)d_in[12];
    const float* nfw     = (const float*)d_in[13];

    const int L = MAMBA_L, DM = MAMBA_DM, DI = MAMBA_DI, DX = MAMBA_DX;
    const int NX = MAMBA_NX, DTR = MAMBA_DTR, VOC = MAMBA_VOC;

    float* ws   = (float*)d_ws;
    float* x    = ws;  ws += (size_t)L * DM;   // residual stream
    float* xn   = ws;  ws += (size_t)L * DM;   // rmsnorm output
    float* xz   = ws;  ws += (size_t)L * DX;   // in_proj output (xi | z)
    float* xc   = ws;  ws += (size_t)L * DI;   // conv+silu output
    float* xdbl = ws;  ws += (size_t)L * NX;   // x_proj output (dt|B|C)
    float* dtb  = ws;  ws += (size_t)L * DI;   // softplus(dt)
    float* yb   = ws;  ws += (size_t)L * DI;   // scan output (gated)
    float* nll  = ws;  ws += (size_t)L;        // per-row nll

    float* logits = (float*)d_out;
    float* loss   = logits + (size_t)L * VOC;

    auto launch_gemm = [&](const float* A, int lda, const float* W,
                           const float* bias, const float* res, float* C,
                           int M, int N, int K, int act) {
        const int tilesM = (M + BLK_M - 1) / BLK_M;
        const int tilesN = (N + BLK_N - 1) / BLK_N;
        gemm_bf16_wmma<<<dim3(tilesM * tilesN), dim3(256), 0, stream>>>(
            A, lda, W, bias, res, C, M, N, K, act);
    };

    // embedding gather
    embed_kernel<<<dim3(L), dim3(256), 0, stream>>>(idx, emb, x, DM);

    for (int l = 0; l < MAMBA_NL; ++l) {
        const float* nw  = norm_w + (size_t)l * DM;
        const float* iw  = in_w   + (size_t)l * DX * DM;
        const float* cw  = conv_w + (size_t)l * DI * 4;
        const float* cb  = conv_b + (size_t)l * DI;
        const float* xw  = xp_w   + (size_t)l * NX * DI;
        const float* dw  = dt_w   + (size_t)l * DI * DTR;
        const float* db  = dt_b   + (size_t)l * DI;
        const float* al  = a_log  + (size_t)l * DI * MAMBA_DST;
        const float* dp  = d_par  + (size_t)l * DI;
        const float* ow  = out_w  + (size_t)l * DM * DI;

        // rmsnorm
        rmsnorm_kernel<<<dim3(L), dim3(256), 0, stream>>>(x, nw, xn, DM);
        // in_proj: (L x DM) @ (DX x DM)^T -> xz (L x DX)
        launch_gemm(xn, DM, iw, nullptr, nullptr, xz, L, DX, DM, 0);
        // causal conv + silu on xi half of xz -> xc
        {
            const int tot = L * DI;
            conv_silu_kernel<<<dim3((tot + 255) / 256), dim3(256), 0, stream>>>(
                xz, cw, cb, xc, L, DI, DX);
        }
        // x_proj: (L x DI) @ (NX x DI)^T -> xdbl (L x NX)
        launch_gemm(xc, DI, xw, nullptr, nullptr, xdbl, L, NX, DI, 0);
        // dt_proj + softplus: (L x DTR slice of xdbl) @ (DI x DTR)^T -> dtb
        launch_gemm(xdbl, NX, dw, db, nullptr, dtb, L, DI, DTR, 1);
        // selective scan + z-gate -> yb
        scan_kernel<<<dim3((DI + 255) / 256), dim3(256), 0, stream>>>(
            xc, dtb, xdbl, al, dp, xz, yb, L, DI, NX, DX);
        // out_proj + residual: (L x DI) @ (DM x DI)^T + x -> x
        launch_gemm(yb, DI, ow, nullptr, x, x, L, DM, DI, 0);
    }

    // final rmsnorm
    rmsnorm_kernel<<<dim3(L), dim3(256), 0, stream>>>(x, nfw, xn, DM);
    // logits: (L x DM) @ (VOC x DM)^T -> d_out
    launch_gemm(xn, DM, emb, nullptr, nullptr, logits, L, VOC, DM, 0);
    // per-row nll + masked mean loss
    nll_kernel<<<dim3(L), dim3(256), 0, stream>>>(logits, tgt, nll, VOC);
    loss_kernel<<<dim3(1), dim3(256), 0, stream>>>(nll, tgt, loss, L);
}
